// MultiHeadLatentAttention_89515708383265
// MI455X (gfx1250) — compile-verified, matrix operand order checked by reading, not run on hardware
//
#include <hip/hip_runtime.h>
#include <hip/hip_bf16.h>

#define DIM 2048
#define N_HEADS 16
#define Q_LORA 1536
#define KV_LORA 512
#define NOPE 128
#define ROPE_D 64
#define V_DIM 128
#define QK_HEAD 192
#define SEQ 4096
#define EPS 1e-6f
#define ATT_SCALE 0.07216878364870322f   /* 192^-0.5 */
#define LOG_THETA 9.210340371976184f     /* ln(10000) */

typedef __bf16 bf16_t;
typedef bf16_t v16bf __attribute__((ext_vector_type(16)));
typedef float  v8f   __attribute__((ext_vector_type(8)));
typedef unsigned int u32x4 __attribute__((ext_vector_type(4)));

struct frag_raw { u32x4 a, b; };

__device__ __forceinline__ unsigned short f2bf(float f) {
    unsigned int u = __float_as_uint(f);
    u += 0x7FFFu + ((u >> 16) & 1u);          // round-to-nearest-even
    return (unsigned short)(u >> 16);
}

// Load one 16x32 bf16 A/B WMMA fragment for this lane.
// p points at &M[row][kbase + kgroup]; elements 0..7 = p[0..7], 8..15 = p[16..23].
__device__ __forceinline__ v16bf load_frag16(const unsigned short* p) {
    frag_raw r;
    r.a = *(const u32x4*)(p);
    r.b = *(const u32x4*)(p + 16);
    return __builtin_bit_cast(v16bf, r);
}

__device__ __forceinline__ v8f wmma_bf16f32(v16bf a, v16bf b, v8f c) {
    return __builtin_amdgcn_wmma_f32_16x16x32_bf16(false, a, false, b,
                                                   (short)0, c, false, false);
}

// CDNA5 async DMA: global -> LDS, 16 bytes per lane, tracked by ASYNCcnt.
__device__ __forceinline__ void async_g2l_b128(unsigned lds_off, const void* gaddr) {
    asm volatile("global_load_async_to_lds_b128 %0, %1, off"
                 :: "v"(lds_off), "v"(gaddr) : "memory");
}
__device__ __forceinline__ void wait_async0() {
    asm volatile("s_wait_asynccnt 0x0" ::: "memory");
}
__device__ __forceinline__ unsigned lds_off32(const void* p) {
    return (unsigned)(unsigned long long)p;   // low 32 bits of generic ptr == LDS offset
}

// ---------------------------------------------------------------------------
// Generic batched GEMM: C[M,N] = A[M,K] @ B[N,K]^T  (bf16 in, f32 accum)
// grid = (N/64, M/128, batch), block = 128 (4 waves).
// WG tile 128x64; A(128x32) + B(64x32) double-buffered in LDS via async DMA.
// Each wave computes a 32x64 slab: 8 WMMAs per 32-wide K step.
// ---------------------------------------------------------------------------
template <bool BF16OUT>
__global__ __launch_bounds__(128) void gemm_abt(
    const unsigned short* __restrict__ A, int lda, long long sA,
    const unsigned short* __restrict__ B, int ldb, long long sB,
    void* __restrict__ Cv, int ldc, long long sC,
    int K)
{
    __shared__ __align__(16) unsigned short At[2][128 * 32];
    __shared__ __align__(16) unsigned short Bt[2][64 * 32];

    const int tid  = threadIdx.x;
    const int w    = tid >> 5;
    const int lane = tid & 31;
    const int lr   = lane & 15;
    const int lk   = (lane >> 4) * 8;

    const size_t rowtile = (size_t)blockIdx.y * 128;
    const size_t coltile = (size_t)blockIdx.x * 64;

    const unsigned short* Ab = A + (size_t)blockIdx.z * sA;
    const unsigned short* Bb = B + (size_t)blockIdx.z * sB;

    auto issue = [&](int buf, int k) {
#pragma unroll
        for (int i = 0; i < 4; ++i) {           // A tile: 512 x 16B chunks / 128 thr
            int c = tid * 4 + i;
            int row = c >> 2, koff = (c & 3) * 8;
            const unsigned short* ga = Ab + (rowtile + row) * (size_t)lda + k + koff;
            async_g2l_b128(lds_off32(&At[buf][row * 32 + koff]), ga);
        }
#pragma unroll
        for (int i = 0; i < 2; ++i) {           // B tile: 256 x 16B chunks / 128 thr
            int c = tid * 2 + i;
            int row = c >> 2, koff = (c & 3) * 8;
            const unsigned short* gb = Bb + (coltile + row) * (size_t)ldb + k + koff;
            async_g2l_b128(lds_off32(&Bt[buf][row * 32 + koff]), gb);
        }
    };

    v8f acc[2][4] = {};

    issue(0, 0);
    wait_async0();
    __syncthreads();

    const int nk = K >> 5;
    for (int kb = 0; kb < nk; ++kb) {
        const int cur = kb & 1;
        if (kb + 1 < nk) issue(cur ^ 1, (kb + 1) << 5);   // prefetch next tile pair

        const unsigned short* at = &At[cur][0];
        const unsigned short* bt = &Bt[cur][0];
        v16bf af0 = load_frag16(at + (w * 32 +  0 + lr) * 32 + lk);
        v16bf af1 = load_frag16(at + (w * 32 + 16 + lr) * 32 + lk);
        v16bf bfr[4];
#pragma unroll
        for (int sn = 0; sn < 4; ++sn)
            bfr[sn] = load_frag16(bt + (sn * 16 + lr) * 32 + lk);
#pragma unroll
        for (int sn = 0; sn < 4; ++sn) {
            acc[0][sn] = wmma_bf16f32(af0, bfr[sn], acc[0][sn]);
            acc[1][sn] = wmma_bf16f32(af1, bfr[sn], acc[1][sn]);
        }

        if (kb + 1 < nk) wait_async0();          // next buffer landed
        __syncthreads();                         // all waves done reading cur
    }

    const size_t bz = (size_t)blockIdx.z * (size_t)sC;
#pragma unroll
    for (int sm = 0; sm < 2; ++sm)
#pragma unroll
        for (int sn = 0; sn < 4; ++sn)
#pragma unroll
            for (int v = 0; v < 8; ++v) {
                size_t r = rowtile + w * 32 + sm * 16 + (lane >> 4) * 8 + v;
                size_t c = coltile + sn * 16 + lr;
                size_t idx = bz + r * (size_t)ldc + c;
                if (BF16OUT) ((unsigned short*)Cv)[idx] = f2bf(acc[sm][sn][v]);
                else         ((float*)Cv)[idx]          = acc[sm][sn][v];
            }
}

// ---------------------------------------------------------------------------
// Flash-style causal MLA attention.
// grid = (SEQ/32, N_HEADS), block = 256 (8 waves).
// q_combined [SEQ][16][576] bf16, k_combined [SEQ][576] bf16,
// kvT [512][SEQ] bf16 (V transposed), ctx [SEQ][16][512] bf16 out.
// ---------------------------------------------------------------------------
__global__ __launch_bounds__(256) void mla_attn(
    const unsigned short* __restrict__ qcomb,
    const unsigned short* __restrict__ kcomb,
    const unsigned short* __restrict__ kvT,
    unsigned short* __restrict__ ctx)
{
    __shared__ float S[32][68];                             // padded vs bank conflicts
    __shared__ __align__(16) unsigned short P[32][72];      // row stride 144B (16B mult)
    __shared__ float m_s[32], l_s[32], a_s[32];

    const int tid  = threadIdx.x;
    const int w    = tid >> 5;
    const int lane = tid & 31;
    const int lr   = lane & 15;
    const int lk   = (lane >> 4) * 8;
    const int qb   = blockIdx.x;
    const int head = blockIdx.y;
    const int qbase = qb * 32;

    const int mi = w & 1;    // phase-A tile row (0..1)
    const int ni = w >> 1;   // phase-A tile col (0..3)
    const int cb = w * 64;   // phase-C output columns owned by this wave

    if (tid < 32) { m_s[tid] = -1e30f; l_s[tid] = 0.0f; }
    __syncthreads();

    v8f O[2][4] = {};

    const unsigned short* qrow =
        qcomb + ((size_t)(qbase + mi * 16 + lr) * N_HEADS + head) * 576 + lk;

    const int nkb = (qbase + 32 + 63) >> 6;
    for (int kb = 0; kb < nkb; ++kb) {
        const int kvbase = kb * 64;

        // ---- Phase A: S = scale * Q' K'^T over 576-dim latent key ----
        v8f sacc = {};
        const unsigned short* krow =
            kcomb + (size_t)(kvbase + ni * 16 + lr) * 576 + lk;
        __builtin_prefetch(krow + 64 * 576, 0, 0);          // next kv block
#pragma unroll 3
        for (int kc = 0; kc < 576; kc += 32) {
            v16bf a = load_frag16(qrow + kc);
            v16bf b = load_frag16(krow + kc);
            sacc = wmma_bf16f32(a, b, sacc);
        }
#pragma unroll
        for (int v = 0; v < 8; ++v) {
            int r = mi * 16 + (lane >> 4) * 8 + v;
            int c = ni * 16 + lr;
            float val = sacc[v] * ATT_SCALE;
            if (kvbase + c > qbase + r) val = -1e30f;       // causal mask
            S[r][c] = val;
        }
        __syncthreads();

        // ---- Phase B: online softmax row update ----
        if (tid < 32) {
            int r = tid;
            float mold = m_s[r], mnew = mold;
            for (int c = 0; c < 64; ++c) mnew = fmaxf(mnew, S[r][c]);
            float alpha = __expf(mold - mnew);
            float sum = 0.0f;
            for (int c = 0; c < 64; ++c) {
                float pe = __expf(S[r][c] - mnew);
                P[r][c] = f2bf(pe);
                sum += pe;
            }
            l_s[r] = l_s[r] * alpha + sum;
            m_s[r] = mnew;
            a_s[r] = alpha;
        }
        __syncthreads();

        // ---- Phase C: O = O*alpha + P @ V (each wave: 64 of 512 cols) ----
#pragma unroll
        for (int fm = 0; fm < 2; ++fm) {
            const unsigned short* prow = &P[fm * 16 + lr][0];
            v16bf pa0 = load_frag16(prow + 0  + lk);
            v16bf pa1 = load_frag16(prow + 32 + lk);
#pragma unroll
            for (int fn = 0; fn < 4; ++fn) {
#pragma unroll
                for (int v = 0; v < 8; ++v)
                    O[fm][fn][v] *= a_s[fm * 16 + (lane >> 4) * 8 + v];
                const unsigned short* vrow =
                    kvT + (size_t)(cb + fn * 16 + lr) * SEQ + kvbase + lk;
                v16bf b0 = load_frag16(vrow + 0);
                v16bf b1 = load_frag16(vrow + 32);
                O[fm][fn] = wmma_bf16f32(pa0, b0, O[fm][fn]);
                O[fm][fn] = wmma_bf16f32(pa1, b1, O[fm][fn]);
            }
        }
        __syncthreads();
    }

    // ---- Epilogue: normalize and store ctx as bf16 ----
#pragma unroll
    for (int fm = 0; fm < 2; ++fm)
#pragma unroll
        for (int fn = 0; fn < 4; ++fn)
#pragma unroll
            for (int v = 0; v < 8; ++v) {
                int r = fm * 16 + (lane >> 4) * 8 + v;
                float val = O[fm][fn][v] / l_s[r];
                int s = qbase + r;
                int c = cb + fn * 16 + lr;
                ctx[((size_t)s * N_HEADS + head) * 512 + c] = f2bf(val);
            }
}

// ---------------------------------------------------------------------------
// Helper kernels
// ---------------------------------------------------------------------------
__global__ void rope_tab(float* __restrict__ cost, float* __restrict__ sint) {
    int s = blockIdx.x, i = threadIdx.x;                    // 32 freqs
    float inv = __expf(-(2.0f * (float)i / (float)ROPE_D) * LOG_THETA);
    float f = (float)s * inv;
    cost[s * 32 + i] = __cosf(f);
    sint[s * 32 + i] = __sinf(f);
}

__global__ void f32_to_bf16_k(const float* __restrict__ in,
                              unsigned short* __restrict__ out, long long n) {
    long long i = (long long)blockIdx.x * blockDim.x + threadIdx.x;
    long long stride = (long long)gridDim.x * blockDim.x;
    for (; i < n; i += stride) out[i] = f2bf(in[i]);
}

__global__ __launch_bounds__(128) void wkvb_nope_transpose(
    const float* __restrict__ wkvb, unsigned short* __restrict__ outT) {
    int c = blockIdx.x;   // 0..511
    int h = blockIdx.y;   // 0..15
    int d = threadIdx.x;  // 0..127
    outT[((size_t)h * 512 + c) * 128 + d] =
        f2bf(wkvb[((size_t)h * 256 + d) * 512 + c]);
}

__global__ __launch_bounds__(256) void rmsnorm_q(
    const float* __restrict__ t1, const float* __restrict__ wgt,
    unsigned short* __restrict__ outb) {
    __shared__ float red[256];
    __shared__ float rs_s;
    int row = blockIdx.x;
    const float* x = t1 + (size_t)row * Q_LORA;
    float ss = 0.0f;
    for (int c = threadIdx.x; c < Q_LORA; c += 256) { float v = x[c]; ss += v * v; }
    red[threadIdx.x] = ss;
    __syncthreads();
    for (int s2 = 128; s2 > 0; s2 >>= 1) {
        if (threadIdx.x < s2) red[threadIdx.x] += red[threadIdx.x + s2];
        __syncthreads();
    }
    if (threadIdx.x == 0) rs_s = rsqrtf(red[0] / (float)Q_LORA + EPS);
    __syncthreads();
    float rs = rs_s;
    for (int c = threadIdx.x; c < Q_LORA; c += 256)
        outb[(size_t)row * Q_LORA + c] = f2bf(x[c] * rs * wgt[c]);
}

__global__ __launch_bounds__(256) void kv_prep(
    const float* __restrict__ kva, const float* __restrict__ wgt,
    const float* __restrict__ cost, const float* __restrict__ sint,
    unsigned short* __restrict__ kcomb, unsigned short* __restrict__ kvT) {
    __shared__ float red[256];
    __shared__ float rs_s;
    int s = blockIdx.x;
    const float* x = kva + (size_t)s * 576;
    float ss = 0.0f;
    for (int c = threadIdx.x; c < KV_LORA; c += 256) { float v = x[c]; ss += v * v; }
    red[threadIdx.x] = ss;
    __syncthreads();
    for (int s2 = 128; s2 > 0; s2 >>= 1) {
        if (threadIdx.x < s2) red[threadIdx.x] += red[threadIdx.x + s2];
        __syncthreads();
    }
    if (threadIdx.x == 0) rs_s = rsqrtf(red[0] / (float)KV_LORA + EPS);
    __syncthreads();
    float rs = rs_s;
    for (int c = threadIdx.x; c < KV_LORA; c += 256) {
        unsigned short b = f2bf(x[c] * rs * wgt[c]);
        kcomb[(size_t)s * 576 + c] = b;
        kvT[(size_t)c * SEQ + s] = b;                       // transposed V
    }
    if (threadIdx.x < 32) {
        int d = threadIdx.x;
        float x0 = x[512 + 2 * d], x1 = x[512 + 2 * d + 1];
        float cv = cost[s * 32 + d], sv = sint[s * 32 + d];
        kcomb[(size_t)s * 576 + 512 + 2 * d]     = f2bf(x0 * cv - x1 * sv);
        kcomb[(size_t)s * 576 + 512 + 2 * d + 1] = f2bf(x0 * sv + x1 * cv);
    }
}

__global__ __launch_bounds__(128) void q_prep(
    const float* __restrict__ q, const float* __restrict__ cost,
    const float* __restrict__ sint, unsigned short* __restrict__ qnope,
    unsigned short* __restrict__ qcomb) {
    int s = blockIdx.x, h = blockIdx.y, d = threadIdx.x;
    const float* qr = q + (size_t)s * (N_HEADS * QK_HEAD) + (size_t)h * QK_HEAD;
    qnope[((size_t)s * N_HEADS + h) * 128 + d] = f2bf(qr[d]);
    if (d < 32) {
        float x0 = qr[128 + 2 * d], x1 = qr[128 + 2 * d + 1];
        float cv = cost[s * 32 + d], sv = sint[s * 32 + d];
        size_t o = ((size_t)s * N_HEADS + h) * 576 + 512;
        qcomb[o + 2 * d]     = f2bf(x0 * cv - x1 * sv);
        qcomb[o + 2 * d + 1] = f2bf(x0 * sv + x1 * cv);
    }
}

// ---------------------------------------------------------------------------
extern "C" void kernel_launch(void* const* d_in, const int* in_sizes, int n_in,
                              void* d_out, int out_size, void* d_ws, size_t ws_size,
                              hipStream_t stream) {
    const float* x       = (const float*)d_in[0];
    const float* wq_a    = (const float*)d_in[1];
    const float* q_norm  = (const float*)d_in[2];
    const float* wq_b    = (const float*)d_in[3];
    const float* wkv_a   = (const float*)d_in[4];
    const float* kv_norm = (const float*)d_in[5];
    const float* wkv_b   = (const float*)d_in[6];
    const float* wo      = (const float*)d_in[7];
    float* out           = (float*)d_out;

    char* p = (char*)d_ws;
    auto take = [&](size_t elems, size_t esz) {
        void* r = (void*)p;
        p += ((elems * esz + 255) & ~(size_t)255);
        return r;
    };

    float* cost            = (float*)take((size_t)SEQ * 32, 4);
    float* sint            = (float*)take((size_t)SEQ * 32, 4);
    unsigned short* x_bf   = (unsigned short*)take((size_t)SEQ * DIM, 2);
    unsigned short* wqa_bf = (unsigned short*)take((size_t)Q_LORA * DIM, 2);
    unsigned short* wqb_bf = (unsigned short*)take((size_t)N_HEADS * QK_HEAD * Q_LORA, 2);
    unsigned short* wkva_bf= (unsigned short*)take((size_t)(KV_LORA + ROPE_D) * DIM, 2);
    unsigned short* wkvb_bf= (unsigned short*)take((size_t)N_HEADS * 256 * 512, 2);
    unsigned short* wo_bf  = (unsigned short*)take((size_t)DIM * DIM, 2);
    unsigned short* wkvbT  = (unsigned short*)take((size_t)N_HEADS * 512 * 128, 2);
    float* t1              = (float*)take((size_t)SEQ * Q_LORA, 4);
    unsigned short* qa_bf  = (unsigned short*)take((size_t)SEQ * Q_LORA, 2);
    float* qf              = (float*)take((size_t)SEQ * N_HEADS * QK_HEAD, 4);
    unsigned short* qnope  = (unsigned short*)take((size_t)SEQ * N_HEADS * 128, 2);
    unsigned short* qcomb  = (unsigned short*)take((size_t)SEQ * N_HEADS * 576, 2);
    float* kva             = (float*)take((size_t)SEQ * 576, 4);
    unsigned short* kcomb  = (unsigned short*)take((size_t)SEQ * 576, 2);
    unsigned short* kvT    = (unsigned short*)take((size_t)512 * SEQ, 2);
    unsigned short* ctx    = (unsigned short*)take((size_t)SEQ * N_HEADS * 512, 2);
    unsigned short* v_bf   = (unsigned short*)take((size_t)SEQ * DIM, 2);

    // Precompute + convert
    rope_tab<<<SEQ, 32, 0, stream>>>(cost, sint);
    f32_to_bf16_k<<<2048, 256, 0, stream>>>(x, x_bf, (long long)SEQ * DIM);
    f32_to_bf16_k<<<2048, 256, 0, stream>>>(wq_a, wqa_bf, (long long)Q_LORA * DIM);
    f32_to_bf16_k<<<2048, 256, 0, stream>>>(wq_b, wqb_bf, (long long)N_HEADS * QK_HEAD * Q_LORA);
    f32_to_bf16_k<<<2048, 256, 0, stream>>>(wkv_a, wkva_bf, (long long)(KV_LORA + ROPE_D) * DIM);
    f32_to_bf16_k<<<2048, 256, 0, stream>>>(wkv_b, wkvb_bf, (long long)N_HEADS * 256 * 512);
    f32_to_bf16_k<<<2048, 256, 0, stream>>>(wo, wo_bf, (long long)DIM * DIM);
    wkvb_nope_transpose<<<dim3(512, 16), 128, 0, stream>>>(wkv_b, wkvbT);

    // t1 = x @ wq_a^T  (4096 x 1536)
    gemm_abt<false><<<dim3(Q_LORA / 64, SEQ / 128, 1), 128, 0, stream>>>(
        x_bf, DIM, 0, wqa_bf, DIM, 0, t1, Q_LORA, 0, DIM);
    rmsnorm_q<<<SEQ, 256, 0, stream>>>(t1, q_norm, qa_bf);

    // q = qa @ wq_b^T  (4096 x 3072)
    gemm_abt<false><<<dim3((N_HEADS * QK_HEAD) / 64, SEQ / 128, 1), 128, 0, stream>>>(
        qa_bf, Q_LORA, 0, wqb_bf, Q_LORA, 0, qf, N_HEADS * QK_HEAD, 0, Q_LORA);

    // kv_a = x @ wkv_a^T  (4096 x 576)
    gemm_abt<false><<<dim3(576 / 64, SEQ / 128, 1), 128, 0, stream>>>(
        x_bf, DIM, 0, wkva_bf, DIM, 0, kva, 576, 0, DIM);

    kv_prep<<<SEQ, 256, 0, stream>>>(kva, kv_norm, cost, sint, kcomb, kvT);
    q_prep<<<dim3(SEQ, N_HEADS), 128, 0, stream>>>(qf, cost, sint, qnope, qcomb);

    // q_abs per head: q_combined[:, h, 0:512] = q_nope[h] @ wkvbT[h]^T
    gemm_abt<true><<<dim3(512 / 64, SEQ / 128, N_HEADS), 128, 0, stream>>>(
        qnope, N_HEADS * 128, 128,
        wkvbT, 128, (long long)512 * 128,
        qcomb, N_HEADS * 576, 576, 128);

    // Flash causal attention in latent space
    mla_attn<<<dim3(SEQ / 32, N_HEADS), 256, 0, stream>>>(qcomb, kcomb, kvT, ctx);

    // v per head: v_bf[:, h*128:(h+1)*128] = ctx[h] @ wkv_b_v[h]^T
    gemm_abt<true><<<dim3(V_DIM / 64, SEQ / 128, N_HEADS), 128, 0, stream>>>(
        ctx, N_HEADS * 512, 512,
        wkvb_bf + 128 * 512, 512, (long long)256 * 512,
        v_bf, DIM, 128, 512);

    // out = v @ wo^T  (f32)
    gemm_abt<false><<<dim3(DIM / 64, SEQ / 128, 1), 128, 0, stream>>>(
        v_bf, DIM, 0, wo_bf, DIM, 0, out, DIM, 0, DIM);
}
